// SlidingKernelAttention_15625091023497
// MI455X (gfx1250) — matrix operand order repaired and checked by reading, not compiled, not run on hardware
//
#include <hip/hip_runtime.h>
#include <hip/hip_bf16.h>
#include <cstdint>

// Problem constants (from reference)
#define B_      4
#define L_      4096
#define C_      1024
#define HEADS_  16
#define HD_     64
#define W_      2047            // (L - 4)/2 + 1
#define M_      (B_ * L_)       // 16384 rows
#define N1_     (3 * C_)        // 3072
#define KDIM_   C_              // 1024

typedef __attribute__((ext_vector_type(16))) __bf16 v16bf;
typedef __attribute__((ext_vector_type(8)))  float  v8f;

// ---------------- bf16 split helpers (bit-exact, RNE) ----------------
__device__ __forceinline__ uint16_t f2bf(float f) {
    uint32_t u = __float_as_uint(f);
    uint32_t r = u + 0x7fffu + ((u >> 16) & 1u);
    return (uint16_t)(r >> 16);
}
__device__ __forceinline__ float bf2f(uint16_t h) {
    return __uint_as_float(((uint32_t)h) << 16);
}

// ---------------- CDNA5 async global->LDS (ASYNCcnt path) ----------------
__device__ __forceinline__ void async_b128(uint32_t ldsOff, const void* g) {
    asm volatile("global_load_async_to_lds_b128 %0, %1, off"
                 :: "v"(ldsOff), "v"((unsigned long long)(uintptr_t)g)
                 : "memory");
}
// Async loads complete in order within a wave: waiting to <=8 releases the
// older 8-load group while the newer group is still in flight.
__device__ __forceinline__ void wait_async_le8() {
    asm volatile("s_wait_asynccnt 0x8" ::: "memory");
}
__device__ __forceinline__ void wait_async_le0() {
    asm volatile("s_wait_asynccnt 0x0" ::: "memory");
}

// cnt(l): number of (w,j) with 2w + j == l, 0<=w<W_, 0<=j<4
__device__ __forceinline__ int window_count(int l) {
    int c = 0;
    int j = l & 1;
    int w = (l - j) >> 1;
    if (w < W_) ++c;
    w = (l - j - 2) >> 1;
    if (w >= 0 && w < W_) ++c;
    return c;
}

// ---------------- WMMA fragment loads from LDS (ISA 7.12.2 layouts) ----------
constexpr int TLD  = 40;           // padded row stride: 80B = 20 banks, 16B aligned
constexpr int SEG  = 128 * TLD;    // elements per LDS array
constexpr int SEGB = SEG * 2;      // bytes per LDS array (10240)
constexpr int BUFE = 4 * SEG;      // elements per double-buffer stage
constexpr int BUFB = BUFE * 2;     // bytes per stage (40960)
constexpr unsigned GEMM_LDS_BYTES = 2 * BUFB;  // 81920 (dynamic LDS; WGP has 320KB)

// A (16x32 bf16): lane (g=lane>>4, m=lane&15) holds row m.
// Element pairs i=0..3 are K = 8g + {0..7}  -> bytes row*80 + 16g + 0..15
// Element pairs i=4..7 are K = 16 + 8g + {0..7} -> bytes row*80 + 16g + 32..47
// => two 16B-aligned contiguous runs: 2x ds_load_b128, conflict-free (20m+4g banks).
__device__ __forceinline__ v16bf load_a_frag(const uint16_t* s, int rowBase, int g, int ln) {
    const uint16_t* rp = s + (rowBase + ln) * TLD + 8 * g;
    rp = (const uint16_t*)__builtin_assume_aligned(rp, 16);
    union { v16bf v; uint4 q[2]; } r;
    r.q[0] = *(const uint4*)rp;
    r.q[1] = *(const uint4*)(rp + 16);
    return r.v;
}

// B (32x16 bf16): lane (g, n) holds column n, K = 8g + {0..7} and 16 + 8g + {0..7}.
// Bs holds B pre-transposed (N x K): runs = bytes row*80 + 16g + 0..15 and 16g + 32..47
// => two 16B-aligned contiguous runs: 2x ds_load_b128.
__device__ __forceinline__ v16bf load_b_frag(const uint16_t* s, int colBase, int g, int ln) {
    const uint16_t* rp = s + (colBase + ln) * TLD + 8 * g;
    rp = (const uint16_t*)__builtin_assume_aligned(rp, 16);
    union { v16bf v; uint4 q[2]; } r;
    r.q[0] = *(const uint4*)rp;
    r.q[1] = *(const uint4*)(rp + 16);
    return r.v;
}

// Stage one K=32 slice of all four operand tiles (Ah/Al/Bh/Bl) into one buffer.
__device__ __forceinline__ void issue_tile(const uint16_t* pAh, const uint16_t* pAl,
                                           const uint16_t* pBh, const uint16_t* pBl,
                                           uint32_t dst) {
    async_b128(dst,                 pAh);
    async_b128(dst + 16,            pAh + 8);
    async_b128(dst + SEGB,          pAl);
    async_b128(dst + SEGB + 16,     pAl + 8);
    async_b128(dst + 2 * SEGB,      pBh);
    async_b128(dst + 2 * SEGB + 16, pBh + 8);
    async_b128(dst + 3 * SEGB,      pBl);
    async_b128(dst + 3 * SEGB + 16, pBl + 8);
}

// ---------------- bf16x3 split-precision GEMM --------------------------------
// Out[M,N] (fp32) = A[M,K] * B[K,N], A given as hi/lo bf16 (row-major),
// B given as hi/lo bf16 transposed (BT: N x K row-major).
// BIAS: Out[row][n] += window_count(row % L) * bias[n]
template <int BIAS>
__global__ __launch_bounds__(256)
void gemm_bf16x3(const uint16_t* __restrict__ Ah, const uint16_t* __restrict__ Al,
                 const uint16_t* __restrict__ Bh, const uint16_t* __restrict__ Bl,
                 float* __restrict__ Out, const float* __restrict__ bias,
                 int N, int K)
{
    extern __shared__ __align__(16) uint16_t smem[];   // 2 stages x [AsH|AsL|BsH|BsL]

    const int tid  = threadIdx.x;
    const int lane = tid & 31;
    const int wid  = tid >> 5;
    const int wm   = (wid & 3) * 32;   // wave M offset within 128 tile (4 waves)
    const int wn   = (wid >> 2) * 64;  // wave N offset within 128 tile (2 waves)
    const int g    = lane >> 4;
    const int ln   = lane & 15;
    const int m0   = blockIdx.x * 128;
    const int n0   = blockIdx.y * 128;

    // Staging: each thread owns one 16-element (32B) run: row r, k-half hh
    const int r  = tid >> 1;
    const int hh = tid & 1;
    const size_t gA = (size_t)(m0 + r) * K + 16 * hh;
    const size_t gB = (size_t)(n0 + r) * K + 16 * hh;
    const uint32_t dstRow = (uint32_t)(uintptr_t)smem + (uint32_t)(r * TLD + 16 * hh) * 2;

    v8f acc[2][4];
#pragma unroll
    for (int mt = 0; mt < 2; ++mt)
#pragma unroll
        for (int nt = 0; nt < 4; ++nt)
            acc[mt][nt] = (v8f){};

    const int nk = K / 32;
    // Prologue: stage k-slice 0 into buffer 0
    issue_tile(Ah + gA, Al + gA, Bh + gB, Bl + gB, dstRow);

    for (int ko = 0; ko < nk; ++ko) {
        const int cur = ko & 1;
        if (ko + 1 < nk) {
            const int k1 = (ko + 1) * 32;
            issue_tile(Ah + gA + k1, Al + gA + k1, Bh + gB + k1, Bl + gB + k1,
                       dstRow + (cur ^ 1) * BUFB);
            wait_async_le8();      // older group (current slice) is in LDS
        } else {
            wait_async_le0();
        }
        __syncthreads();           // all waves' data for slice `ko` visible

        const uint16_t* AsH = smem + cur * BUFE;
        const uint16_t* AsL = AsH + SEG;
        const uint16_t* BsH = AsH + 2 * SEG;
        const uint16_t* BsL = AsH + 3 * SEG;

        v16bf aH[2], aL[2], bH[4], bL[4];
#pragma unroll
        for (int mt = 0; mt < 2; ++mt) {
            aH[mt] = load_a_frag(AsH, wm + mt * 16, g, ln);
            aL[mt] = load_a_frag(AsL, wm + mt * 16, g, ln);
        }
#pragma unroll
        for (int nt = 0; nt < 4; ++nt) {
            bH[nt] = load_b_frag(BsH, wn + nt * 16, g, ln);
            bL[nt] = load_b_frag(BsL, wn + nt * 16, g, ln);
        }
#pragma unroll
        for (int mt = 0; mt < 2; ++mt)
#pragma unroll
            for (int nt = 0; nt < 4; ++nt) {
                acc[mt][nt] = __builtin_amdgcn_wmma_f32_16x16x32_bf16(
                    false, aH[mt], false, bH[nt], (short)0, acc[mt][nt], false, false);
                acc[mt][nt] = __builtin_amdgcn_wmma_f32_16x16x32_bf16(
                    false, aH[mt], false, bL[nt], (short)0, acc[mt][nt], false, false);
                acc[mt][nt] = __builtin_amdgcn_wmma_f32_16x16x32_bf16(
                    false, aL[mt], false, bH[nt], (short)0, acc[mt][nt], false, false);
            }
        __syncthreads();           // reads of buf[cur] done before it is re-staged
    }

    // Epilogue: C/D layout — VGPR j: lanes 0-15 -> M=j, lanes 16-31 -> M=j+8
#pragma unroll
    for (int mt = 0; mt < 2; ++mt) {
        const int rowBase = m0 + wm + mt * 16 + 8 * g;
#pragma unroll
        for (int nt = 0; nt < 4; ++nt) {
            const int col = n0 + wn + nt * 16 + ln;
            float bv = 0.f;
            if (BIAS) bv = bias[col];
#pragma unroll
            for (int j = 0; j < 8; ++j) {
                const int row = rowBase + j;
                float v = acc[mt][nt][j];
                if (BIAS) v += (float)window_count(row & (L_ - 1)) * bv;
                Out[(size_t)row * N + col] = v;
            }
        }
    }
}

// ---------------- fp32 -> bf16 hi/lo splitters ------------------------------
__global__ __launch_bounds__(256)
void split_hi_lo(const float* __restrict__ src, uint16_t* __restrict__ hi,
                 uint16_t* __restrict__ lo, int n)
{
    const int i = blockIdx.x * 256 + threadIdx.x;
    if (i >= n) return;
    const float v = src[i];
    const uint16_t hb = f2bf(v);
    hi[i] = hb;
    lo[i] = f2bf(v - bf2f(hb));
}

// Transpose K x N weights to N x K and split hi/lo
__global__ __launch_bounds__(256)
void transpose_split(const float* __restrict__ w, uint16_t* __restrict__ th,
                     uint16_t* __restrict__ tl, int K, int N)
{
    __shared__ float tile[32][33];
    const int kb = blockIdx.y * 32;
    const int nb = blockIdx.x * 32;
    const int tx = threadIdx.x;   // 0..31
    const int ty = threadIdx.y;   // 0..7
#pragma unroll
    for (int i = ty; i < 32; i += 8)
        tile[i][tx] = w[(size_t)(kb + i) * N + nb + tx];
    __syncthreads();
#pragma unroll
    for (int i = ty; i < 32; i += 8) {
        const float v = tile[tx][i];
        const size_t o = (size_t)(nb + i) * K + kb + tx;
        const uint16_t hb = f2bf(v);
        th[o] = hb;
        tl[o] = f2bf(v - bf2f(hb));
    }
}

// ---------------- per-position attention gather (no atomics) ----------------
// thread = (b, l, h); sums attention outputs of the <=2 windows covering l,
// writes osum directly as bf16 hi/lo for the second GEMM.
__global__ __launch_bounds__(256)
void attn_kernel(const float* __restrict__ qkv, uint16_t* __restrict__ oh,
                 uint16_t* __restrict__ ol)
{
    const int t  = blockIdx.x * 256 + threadIdx.x;
    const int h  = t & (HEADS_ - 1);
    const int bl = t >> 4;                 // b*L + l
    const int l  = bl & (L_ - 1);
    const int bbase = bl & ~(L_ - 1);      // b*L
    const float scale = 0.03125f;          // 1024^-0.5

    const size_t RS = 3 * C_;              // qkv row stride
    const float* q = qkv + (size_t)bl * RS + h * HD_;

    // candidate windows: j = l&1 and j = (l&1)+2
    const int j0 = l & 1;
    int w0 = (l - j0) >> 1;
    int w1 = (l - j0 - 2) >> 1;
    const bool v0 = (w0 < W_);
    const bool v1 = (w1 >= 0 && w1 < W_);
    int wc[2];
    wc[0] = v0 ? w0 : 0;
    wc[1] = v1 ? w1 : 0;
    const float vmul[2] = { v0 ? 1.f : 0.f, v1 ? 1.f : 0.f };

    float att[2][4];
#pragma unroll
    for (int u = 0; u < 2; ++u) {
        const float* kp = qkv + (size_t)(bbase + 2 * wc[u]) * RS + C_ + h * HD_;
        float d0 = 0.f, d1 = 0.f, d2 = 0.f, d3 = 0.f;
        for (int c = 0; c < HD_; c += 4) {
            const float4 qv = *(const float4*)(q + c);
            const float4 k0v = *(const float4*)(kp + c);
            const float4 k1v = *(const float4*)(kp + RS + c);
            const float4 k2v = *(const float4*)(kp + 2 * RS + c);
            const float4 k3v = *(const float4*)(kp + 3 * RS + c);
            d0 += qv.x * k0v.x + qv.y * k0v.y + qv.z * k0v.z + qv.w * k0v.w;
            d1 += qv.x * k1v.x + qv.y * k1v.y + qv.z * k1v.z + qv.w * k1v.w;
            d2 += qv.x * k2v.x + qv.y * k2v.y + qv.z * k2v.z + qv.w * k2v.w;
            d3 += qv.x * k3v.x + qv.y * k3v.y + qv.z * k3v.z + qv.w * k3v.w;
        }
        d0 *= scale; d1 *= scale; d2 *= scale; d3 *= scale;
        const float m = fmaxf(fmaxf(d0, d1), fmaxf(d2, d3));
        const float e0 = __expf(d0 - m), e1 = __expf(d1 - m);
        const float e2 = __expf(d2 - m), e3 = __expf(d3 - m);
        const float inv = vmul[u] / (e0 + e1 + e2 + e3);
        att[u][0] = e0 * inv; att[u][1] = e1 * inv;
        att[u][2] = e2 * inv; att[u][3] = e3 * inv;
    }

    const float* vp0 = qkv + (size_t)(bbase + 2 * wc[0]) * RS + 2 * C_ + h * HD_;
    const float* vp1 = qkv + (size_t)(bbase + 2 * wc[1]) * RS + 2 * C_ + h * HD_;
    const size_t obase = (size_t)bl * C_ + h * HD_;
    for (int c = 0; c < HD_; c += 4) {
        float ax = 0.f, ay = 0.f, az = 0.f, aw = 0.f;
#pragma unroll
        for (int p = 0; p < 4; ++p) {
            const float4 a = *(const float4*)(vp0 + p * RS + c);
            const float4 b = *(const float4*)(vp1 + p * RS + c);
            ax += att[0][p] * a.x + att[1][p] * b.x;
            ay += att[0][p] * a.y + att[1][p] * b.y;
            az += att[0][p] * a.z + att[1][p] * b.z;
            aw += att[0][p] * a.w + att[1][p] * b.w;
        }
        const float vals[4] = { ax, ay, az, aw };
#pragma unroll
        for (int e = 0; e < 4; ++e) {
            const uint16_t hb = f2bf(vals[e]);
            oh[obase + c + e] = hb;
            ol[obase + c + e] = f2bf(vals[e] - bf2f(hb));
        }
    }
}

// ---------------- launcher --------------------------------------------------
extern "C" void kernel_launch(void* const* d_in, const int* in_sizes, int n_in,
                              void* d_out, int out_size, void* d_ws, size_t ws_size,
                              hipStream_t stream)
{
    (void)in_sizes; (void)n_in; (void)out_size; (void)ws_size;
    const float* x     = (const float*)d_in[0];
    const float* w_qkv = (const float*)d_in[1];
    const float* w_out = (const float*)d_in[2];
    const float* b_out = (const float*)d_in[3];
    float* out = (float*)d_out;

    char* ws = (char*)d_ws;
    size_t off = 0;
    float*    qkv = (float*)(ws + off);    off += (size_t)M_ * N1_ * sizeof(float);    // 201 MB
    uint16_t* xh  = (uint16_t*)(ws + off); off += (size_t)M_ * KDIM_ * 2;
    uint16_t* xl  = (uint16_t*)(ws + off); off += (size_t)M_ * KDIM_ * 2;
    uint16_t* wqh = (uint16_t*)(ws + off); off += (size_t)N1_ * KDIM_ * 2;
    uint16_t* wql = (uint16_t*)(ws + off); off += (size_t)N1_ * KDIM_ * 2;
    uint16_t* woh = (uint16_t*)(ws + off); off += (size_t)C_ * KDIM_ * 2;
    uint16_t* wol = (uint16_t*)(ws + off); off += (size_t)C_ * KDIM_ * 2;
    uint16_t* osh = (uint16_t*)(ws + off); off += (size_t)M_ * KDIM_ * 2;
    uint16_t* osl = (uint16_t*)(ws + off); off += (size_t)M_ * KDIM_ * 2;

    // Stage 0: precision split + weight transposes
    split_hi_lo<<<(M_ * KDIM_ + 255) / 256, 256, 0, stream>>>(x, xh, xl, M_ * KDIM_);
    transpose_split<<<dim3(N1_ / 32, KDIM_ / 32), dim3(32, 8), 0, stream>>>(w_qkv, wqh, wql, KDIM_, N1_);
    transpose_split<<<dim3(C_ / 32, KDIM_ / 32), dim3(32, 8), 0, stream>>>(w_out, woh, wol, KDIM_, C_);

    // Stage 1: qkv = x @ w_qkv  (dedup'd over unique rows)
    gemm_bf16x3<0><<<dim3(M_ / 128, N1_ / 128), 256, GEMM_LDS_BYTES, stream>>>(
        xh, xl, wqh, wql, qkv, nullptr, N1_, KDIM_);

    // Stage 2: per-position windowed attention gather -> osum (bf16 hi/lo)
    attn_kernel<<<(M_ * HEADS_) / 256, 256, 0, stream>>>(qkv, osh, osl);

    // Stage 3: out = osum @ w_out + cnt(l) * b_out
    gemm_bf16x3<1><<<dim3(M_ / 128, C_ / 128), 256, GEMM_LDS_BYTES, stream>>>(
        osh, osl, woh, wol, out, b_out, C_, KDIM_);
}
